// LSTMTransformerBlock_21354577396441
// MI455X (gfx1250) — matrix operand order repaired and checked by reading, not compiled, not run on hardware
//
#include <hip/hip_runtime.h>
#include <math.h>

// ---------------------------------------------------------------------------
// LSTM-Transformer block for MI455X (gfx1250).
// All heavy GEMMs (QKV, O, FFN1, FFN2) AND attention (QK^T, P*V) run on
// v_wmma_f32_16x16x32_bf16 (wave32 WMMA, f32 accumulate).
// ---------------------------------------------------------------------------

typedef __attribute__((ext_vector_type(16))) __bf16 v16bf;
typedef __attribute__((ext_vector_type(8)))  float  v8f;

__device__ inline unsigned short f2bf(float f) {
  unsigned int u = __float_as_uint(f);
  u += 0x7fffu + ((u >> 16) & 1u);   // round-to-nearest-even
  return (unsigned short)(u >> 16);
}
__device__ inline float bf2f(unsigned short u) {
  return __uint_as_float(((unsigned int)u) << 16);
}

union BfFrag { unsigned int u32[8]; v16bf v; };

// --------------------------- weight convert + transpose --------------------
__global__ void k_convertT(const float* __restrict__ W,
                           unsigned short* __restrict__ Wt, int K, int N) {
  int i = blockIdx.x * blockDim.x + threadIdx.x;
  if (i >= K * N) return;
  int k = i / N, n = i - k * N;
  Wt[(size_t)n * K + k] = f2bf(W[i]);
}

// --------------------------------- layernorm --------------------------------
__global__ void k_layernorm(const float* __restrict__ x,
                            const float* __restrict__ g,
                            const float* __restrict__ be,
                            float* __restrict__ outf,
                            unsigned short* __restrict__ outb, int D) {
  const int row = blockIdx.x;
  const float* xr = x + (size_t)row * D;
  __shared__ float red[256];
  float s = 0.f;
  for (int i = threadIdx.x; i < D; i += blockDim.x) s += xr[i];
  red[threadIdx.x] = s; __syncthreads();
  for (int off = 128; off > 0; off >>= 1) {
    if ((int)threadIdx.x < off) red[threadIdx.x] += red[threadIdx.x + off];
    __syncthreads();
  }
  float mean = red[0] / (float)D;
  __syncthreads();
  float v = 0.f;
  for (int i = threadIdx.x; i < D; i += blockDim.x) {
    float d = xr[i] - mean; v += d * d;
  }
  red[threadIdx.x] = v; __syncthreads();
  for (int off = 128; off > 0; off >>= 1) {
    if ((int)threadIdx.x < off) red[threadIdx.x] += red[threadIdx.x + off];
    __syncthreads();
  }
  float inv = rsqrtf(red[0] / (float)D + 1e-5f);
  for (int i = threadIdx.x; i < D; i += blockDim.x) {
    float o = (xr[i] - mean) * inv * g[i] + be[i];
    if (outf) outf[(size_t)row * D + i] = o;
    if (outb) outb[(size_t)row * D + i] = f2bf(o);
  }
}

// ------------------------------ WMMA fragments ------------------------------
// A fragment: 16x32 bf16, ISA layout: lane(0-15)=row, lane>>4 selects K-half.
__device__ inline v16bf load_a_frag(const unsigned short* __restrict__ A,
                                    int lda, int row0, int k0, int lane) {
  BfFrag f;
  const int r  = row0 + (lane & 15);
  const int kh = (lane >> 4) << 3;                 // 0 or 8
  const unsigned short* p = A + (size_t)r * lda + k0;
#pragma unroll
  for (int i = 0; i < 8; ++i) {
    int k = ((i >> 2) << 4) + kh + ((i & 3) << 1); // K pattern per ISA table
    f.u32[i] = *(const unsigned int*)(p + k);
  }
  return f.v;
}

// B fragment: 32x16 bf16 from "B transposed" layout [N,K]; lane(0-15)=column,
// lane>>4 selects K-half (same K pattern as A fragment).
__device__ inline v16bf load_b_frag(const unsigned short* __restrict__ Bt,
                                    int ldk, int col0, int k0, int lane) {
  BfFrag f;
  const int c  = col0 + (lane & 15);
  const int kh = (lane >> 4) << 3;                 // 0 or 8
  const unsigned short* p = Bt + (size_t)c * ldk + k0 + kh;
#pragma unroll
  for (int i = 0; i < 8; ++i) f.u32[i] = *(const unsigned int*)(p + ((i >> 2) << 4) + ((i & 3) << 1));
  return f.v;
}

// LDS variants (no __restrict__: must stay ordered with preceding DS stores).
__device__ inline v16bf load_a_frag_lds(const unsigned short* P, int lda, int lane) {
  BfFrag f;
  const int r  = lane & 15;
  const int kh = (lane >> 4) << 3;
  const unsigned short* p = P + r * lda;
#pragma unroll
  for (int i = 0; i < 8; ++i) {
    int k = ((i >> 2) << 4) + kh + ((i & 3) << 1);
    f.u32[i] = *(const unsigned int*)(p + k);
  }
  return f.v;
}
__device__ inline v16bf load_b_frag_lds(const unsigned short* Bt, int ldk,
                                        int col0, int lane) {
  BfFrag f;
  const int c  = col0 + (lane & 15);
  const int kh = (lane >> 4) << 3;
  const unsigned short* p = Bt + c * ldk + kh;
#pragma unroll
  for (int i = 0; i < 8; ++i) f.u32[i] = *(const unsigned int*)(p + ((i >> 2) << 4) + ((i & 3) << 1));
  return f.v;
}

// ------------------------------ WMMA bf16 GEMM ------------------------------
// Block: 128 threads = 4 waves. Each wave owns a 16(M) x 64(N) strip with 4
// accumulators -> one A fragment feeds 4 WMMAs per 32-deep k-step.
__global__ void __launch_bounds__(128)
k_gemm_bf16(const unsigned short* __restrict__ A,   // [M,K] bf16
            const unsigned short* __restrict__ Bt,  // [N,K] bf16 (B transposed)
            const float* __restrict__ bias,         // [N] or null
            const float* __restrict__ resid,        // [M,N] or null
            float* __restrict__ outf,               // [M,N] or null
            unsigned short* __restrict__ outb,      // [M,N] or null
            int M, int N, int K, int act) {
  const int lane = threadIdx.x & 31;
  const int wave = threadIdx.x >> 5;
  const int col0 = blockIdx.x * 256 + wave * 64;   // wave's N strip base
  const int row0 = blockIdx.y * 16;

  v8f acc[4] = {};
  const int arow = row0 + (lane & 15);
  const int bcol = col0 + (lane & 15);
  for (int k0 = 0; k0 < K; k0 += 32) {
    if (k0 + 32 < K) {   // prefetch next k-tile (global_prefetch_b8)
      __builtin_prefetch((const void*)(A + (size_t)arow * K + k0 + 32), 0, 3);
      __builtin_prefetch((const void*)(Bt + (size_t)bcol * K + k0 + 32), 0, 3);
    }
    v16bf a = load_a_frag(A, K, row0, k0, lane);
#pragma unroll
    for (int t = 0; t < 4; ++t) {
      v16bf b = load_b_frag(Bt, K, col0 + 16 * t, k0, lane);
      acc[t] = __builtin_amdgcn_wmma_f32_16x16x32_bf16(
          false, a, false, b, (short)0, acc[t], false, false);
    }
  }

  const int nlo = lane & 15;
  const int mh  = (lane >> 4) << 3;                // row half offset (0 or 8)
#pragma unroll
  for (int t = 0; t < 4; ++t) {
    const int n  = col0 + 16 * t + nlo;
    const float bb = bias ? bias[n] : 0.f;
#pragma unroll
    for (int r = 0; r < 8; ++r) {
      const int m = row0 + mh + r;                 // C layout: VGPR r -> row
      float v = acc[t][r] + bb;
      if (act) v = 0.5f * v * (1.f + erff(v * 0.70710678118654752f)); // gelu
      if (resid) v += resid[(size_t)m * N + n];
      if (outf) outf[(size_t)m * N + n] = v;
      if (outb) outb[(size_t)m * N + n] = f2bf(v);
    }
  }
}

// -------------------------- small f32 GEMM (2 rows) -------------------------
__global__ void k_rowmat(const float* __restrict__ A,  // [R,K]
                         const float* __restrict__ W,  // [K,N]
                         float* __restrict__ out, int R, int K, int N) {
  int i = blockIdx.x * blockDim.x + threadIdx.x;
  if (i >= R * N) return;
  int r = i / N, n = i - r * N;
  float s = 0.f;
  for (int k = 0; k < K; ++k) s += A[(size_t)r * K + k] * W[(size_t)k * N + n];
  out[i] = s;
}

// --------------------------- WMMA flash attention ---------------------------
// Block: 128 threads = 4 waves, each wave one 16-query tile of one (b,h).
// Per 32-key chunk: 4 WMMAs for Q*K^T (K read straight from global; the DH
// contraction is contiguous), online softmax via shfl reductions, P round-
// tripped through per-wave LDS to become an A-fragment, V chunk cooperatively
// staged transposed in LDS; 4 WMMAs for P*V.
__global__ void __launch_bounds__(128)
k_attention_wmma(const unsigned short* __restrict__ Qb,
                 const unsigned short* __restrict__ Kb,
                 const unsigned short* __restrict__ Vb,
                 const float* __restrict__ Km, const float* __restrict__ Vm,
                 unsigned short* __restrict__ ao,
                 int Bn, int S, int H, int DH) {
  const int D    = H * DH;                  // 1024
  const int lane = threadIdx.x & 31;
  const int wave = threadIdx.x >> 5;
  const int h    = blockIdx.y;
  const int b    = blockIdx.z;
  const int q0   = blockIdx.x * 64 + wave * 16;
  const int nlo  = lane & 15;
  const int mh   = lane >> 4;               // 0 or 1 (row half)

  __shared__ unsigned short Vt[64 * 32];    // V^T chunk: [dh][key]
  __shared__ unsigned short Pb[4 * 16 * 32];
  unsigned short* Pl = &Pb[wave * 16 * 32]; // this wave's P tile [q][key]

  const size_t headoff = (size_t)(b * S) * D + (size_t)h * DH;
  const unsigned short* Qhead = Qb + headoff;
  const unsigned short* Khead = Kb + headoff;
  const unsigned short* Vhead = Vb + headoff;

  // Q tile as two A fragments (DH=64 -> k0 = 0, 32)
  v16bf qa0 = load_a_frag(Qhead + (size_t)q0 * D, D, 0, 0, lane);
  v16bf qa1 = load_a_frag(Qhead + (size_t)q0 * D, D, 0, 32, lane);
  const float scale = rsqrtf((float)DH);

  // ---- memory-token prologue: exact init of m, l, O ----
  float mrow[8], lrow[8];
  v8f acc[4];
  {
    const float* km = Km + (size_t)b * D + h * DH;
    const float* vm = Vm + (size_t)b * D + h * DH;
#pragma unroll
    for (int r = 0; r < 8; ++r) {
      int qrow = q0 + mh * 8 + r;
      float s = 0.f;
#pragma unroll
      for (int j = 0; j < 4; ++j) {
        int d = nlo + 16 * j;
        s += bf2f(Qhead[(size_t)qrow * D + d]) * km[d];
      }
      s += __shfl_xor(s, 1, 32);
      s += __shfl_xor(s, 2, 32);
      s += __shfl_xor(s, 4, 32);
      s += __shfl_xor(s, 8, 32);
      mrow[r] = s * scale;
      lrow[r] = 1.f;
    }
#pragma unroll
    for (int t = 0; t < 4; ++t) {
      float v = vm[t * 16 + nlo];
#pragma unroll
      for (int r = 0; r < 8; ++r) acc[t][r] = v;   // weight exp(0)=1
    }
  }

  for (int kc = 0; kc < S; kc += 32) {
    // ---- stage V^T chunk (block cooperative) ----
    __syncthreads();
    for (int i = threadIdx.x; i < 64 * 32; i += 128) {
      int kk = i >> 6, dd = i & 63;                // coalesced read over dd
      Vt[dd * 32 + kk] = Vhead[(size_t)(kc + kk) * D + dd];
    }
    __syncthreads();

    // ---- scores: two 16x16 tiles over keys [kc,kc+16) and [kc+16,kc+32) ----
    v8f s0 = {}, s1 = {};
    {
      v16bf b0 = load_b_frag(Khead, D, kc, 0, lane);
      s0 = __builtin_amdgcn_wmma_f32_16x16x32_bf16(false, qa0, false, b0, (short)0, s0, false, false);
      v16bf b1 = load_b_frag(Khead, D, kc, 32, lane);
      s0 = __builtin_amdgcn_wmma_f32_16x16x32_bf16(false, qa1, false, b1, (short)0, s0, false, false);
      v16bf b2 = load_b_frag(Khead, D, kc + 16, 0, lane);
      s1 = __builtin_amdgcn_wmma_f32_16x16x32_bf16(false, qa0, false, b2, (short)0, s1, false, false);
      v16bf b3 = load_b_frag(Khead, D, kc + 16, 32, lane);
      s1 = __builtin_amdgcn_wmma_f32_16x16x32_bf16(false, qa1, false, b3, (short)0, s1, false, false);
    }

    // ---- online softmax (row stats across 16-lane groups) ----
#pragma unroll
    for (int r = 0; r < 8; ++r) {
      float a0 = s0[r] * scale, a1 = s1[r] * scale;
      float mx = fmaxf(a0, a1);
      mx = fmaxf(mx, __shfl_xor(mx, 1, 32));
      mx = fmaxf(mx, __shfl_xor(mx, 2, 32));
      mx = fmaxf(mx, __shfl_xor(mx, 4, 32));
      mx = fmaxf(mx, __shfl_xor(mx, 8, 32));
      float mnew  = fmaxf(mrow[r], mx);
      float alpha = __expf(mrow[r] - mnew);
      float p0 = __expf(a0 - mnew);
      float p1 = __expf(a1 - mnew);
      float rs = p0 + p1;
      rs += __shfl_xor(rs, 1, 32);
      rs += __shfl_xor(rs, 2, 32);
      rs += __shfl_xor(rs, 4, 32);
      rs += __shfl_xor(rs, 8, 32);
      lrow[r] = lrow[r] * alpha + rs;
      mrow[r] = mnew;
      int row = mh * 8 + r;
      Pl[row * 32 + nlo]      = f2bf(p0);
      Pl[row * 32 + 16 + nlo] = f2bf(p1);
#pragma unroll
      for (int t = 0; t < 4; ++t) acc[t][r] *= alpha;
    }

    // P store -> A-frag load (same wave; keep LDS ordering explicit)
    asm volatile("s_wait_dscnt 0" ::: "memory");

    // ---- P * V (contraction over 32 keys) ----
    v16bf pa = load_a_frag_lds(Pl, 32, lane);
#pragma unroll
    for (int t = 0; t < 4; ++t) {
      v16bf vb = load_b_frag_lds(Vt, 32, t * 16, lane);
      acc[t] = __builtin_amdgcn_wmma_f32_16x16x32_bf16(
          false, pa, false, vb, (short)0, acc[t], false, false);
    }
  }

  // ---- normalize and store (bf16, feeds Wo GEMM) ----
#pragma unroll
  for (int t = 0; t < 4; ++t) {
#pragma unroll
    for (int r = 0; r < 8; ++r) {
      int qrow = q0 + mh * 8 + r;
      ao[headoff + (size_t)qrow * D + t * 16 + nlo] = f2bf(acc[t][r] / lrow[r]);
    }
  }
}

// ------------------------------- mean pooling -------------------------------
__global__ void k_meanpool(const float* __restrict__ att,
                           float* __restrict__ pooled, int Bn, int S, int D) {
  int i = blockIdx.x * blockDim.x + threadIdx.x;
  if (i >= Bn * D) return;
  int b = i / D, d = i - b * D;
  float s = 0.f;
  for (int t = 0; t < S; ++t) s += att[((size_t)b * S + t) * D + d];
  pooled[i] = s / (float)S;
}

// ------------------------------ LSTM gate GEMM ------------------------------
__global__ void k_gates(const float* __restrict__ pooled,
                        const float* __restrict__ h,
                        const float* __restrict__ Wih,
                        const float* __restrict__ Whh,
                        const float* __restrict__ b_ih,
                        const float* __restrict__ b_hh,
                        float* __restrict__ gates, int R, int K, int N4) {
  int i = blockIdx.x * blockDim.x + threadIdx.x;
  if (i >= R * N4) return;
  int r = i / N4, j = i - r * N4;
  float s = b_ih[j] + b_hh[j];
  for (int k = 0; k < K; ++k) s += pooled[(size_t)r * K + k] * Wih[(size_t)k * N4 + j];
  for (int k = 0; k < K; ++k) s += h[(size_t)r * K + k] * Whh[(size_t)k * N4 + j];
  gates[i] = s;
}

// ------------------------------ LSTM cell step ------------------------------
__global__ void k_lstm(const float* __restrict__ gates,
                       const float* __restrict__ c,
                       float* __restrict__ new_h, float* __restrict__ new_c,
                       int R, int DM) {
  int i = blockIdx.x * blockDim.x + threadIdx.x;
  if (i >= R * DM) return;
  int r = i / DM, d = i - r * DM;
  const float* gr = gates + (size_t)r * 4 * DM;
  float gi = gr[d], gf = gr[DM + d], gg = gr[2 * DM + d], go = gr[3 * DM + d];
  float si = 1.f / (1.f + __expf(-gi));
  float sf = 1.f / (1.f + __expf(-gf));
  float so = 1.f / (1.f + __expf(-go));
  float nc = sf * c[i] + si * tanhf(gg);
  float nh = so * tanhf(nc);
  new_c[i] = nc;
  new_h[i] = nh;
}

// -------------------------- broadcast row add -------------------------------
__global__ void k_addrow(const float* __restrict__ a,
                         const float* __restrict__ rowv,
                         float* __restrict__ out, int Bn, int S, int D) {
  size_t total = (size_t)Bn * S * D;
  size_t i = (size_t)blockIdx.x * blockDim.x + threadIdx.x;
  if (i >= total) return;
  int d = (int)(i % D);
  int b = (int)(i / ((size_t)S * D));
  out[i] = a[i] + rowv[(size_t)b * D + d];
}

// ============================================================================
extern "C" void kernel_launch(void* const* d_in, const int* in_sizes, int n_in,
                              void* d_out, int out_size, void* d_ws,
                              size_t ws_size, hipStream_t stream) {
  (void)in_sizes; (void)n_in; (void)out_size; (void)ws_size;
  const int Bn = 2, S = 2048, D = 1024, DM = 1024, H = 16, DH = 64;
  const int Mr = Bn * S;          // 4096 token rows
  const int D4 = 4 * D;           // 4096

  const float* x    = (const float*)d_in[0];
  const float* h    = (const float*)d_in[1];
  const float* c    = (const float*)d_in[2];
  const float* Wq   = (const float*)d_in[3];
  const float* Wk   = (const float*)d_in[4];
  const float* Wv   = (const float*)d_in[5];
  const float* Wo   = (const float*)d_in[6];
  const float* Wmk  = (const float*)d_in[7];
  const float* Wmv  = (const float*)d_in[8];
  const float* Wih  = (const float*)d_in[9];
  const float* Whh  = (const float*)d_in[10];
  const float* b_ih = (const float*)d_in[11];
  const float* b_hh = (const float*)d_in[12];
  const float* W1   = (const float*)d_in[13];
  const float* bf1  = (const float*)d_in[14];
  const float* W2   = (const float*)d_in[15];
  const float* bf2  = (const float*)d_in[16];
  const float* g1   = (const float*)d_in[17];
  const float* be1  = (const float*)d_in[18];
  const float* g2   = (const float*)d_in[19];
  const float* be2  = (const float*)d_in[20];
  const float* g3   = (const float*)d_in[21];
  const float* be3  = (const float*)d_in[22];
  const float* Wmp  = (const float*)d_in[23];

  // -------- workspace carve --------
  char* ws = (char*)d_ws;
  size_t off = 0;
  auto carve = [&](size_t bytes) -> char* {
    char* p = ws + off;
    off += (bytes + 255) & ~(size_t)255;
    return p;
  };
  unsigned short* xn   = (unsigned short*)carve((size_t)Mr * D * 2);
  unsigned short* WqT  = (unsigned short*)carve((size_t)D * D * 2);
  unsigned short* WkT  = (unsigned short*)carve((size_t)D * D * 2);
  unsigned short* WvT  = (unsigned short*)carve((size_t)D * D * 2);
  unsigned short* WoT  = (unsigned short*)carve((size_t)D * D * 2);
  unsigned short* W1T  = (unsigned short*)carve((size_t)D * D4 * 2);
  unsigned short* W2T  = (unsigned short*)carve((size_t)D4 * D * 2);
  unsigned short* Qbuf = (unsigned short*)carve((size_t)Mr * D * 2);
  unsigned short* Kbuf = (unsigned short*)carve((size_t)Mr * D * 2);
  unsigned short* Vbuf = (unsigned short*)carve((size_t)Mr * D * 2);
  float* Kmf           = (float*)carve((size_t)Bn * D * 4);
  float* Vmf           = (float*)carve((size_t)Bn * D * 4);
  unsigned short* aob  = (unsigned short*)carve((size_t)Mr * D * 2);
  float* att           = (float*)carve((size_t)Mr * D * 4);
  float* pooled        = (float*)carve((size_t)Bn * D * 4);
  float* gates         = (float*)carve((size_t)Bn * D4 * 4);
  float* hrow          = (float*)carve((size_t)Bn * D * 4);
  float* mixed         = (float*)carve((size_t)Mr * D * 4);
  unsigned short* hn   = (unsigned short*)carve((size_t)Mr * D * 2);
  unsigned short* ffn1 = (unsigned short*)carve((size_t)Mr * D4 * 2);
  float* y             = (float*)carve((size_t)Mr * D * 4);

  float* out_tok = (float*)d_out;                  // [B,S,D]
  float* out_h   = out_tok + (size_t)Mr * D;       // [B,DM]
  float* out_c   = out_h + (size_t)Bn * DM;        // [B,DM]

  // -------- 1. weight convert+transpose to bf16 --------
  {
    int n1 = D * D;
    k_convertT<<<(n1 + 255) / 256, 256, 0, stream>>>(Wq, WqT, D, D);
    k_convertT<<<(n1 + 255) / 256, 256, 0, stream>>>(Wk, WkT, D, D);
    k_convertT<<<(n1 + 255) / 256, 256, 0, stream>>>(Wv, WvT, D, D);
    k_convertT<<<(n1 + 255) / 256, 256, 0, stream>>>(Wo, WoT, D, D);
    int n2 = D * D4;
    k_convertT<<<(n2 + 255) / 256, 256, 0, stream>>>(W1, W1T, D, D4);
    k_convertT<<<(n2 + 255) / 256, 256, 0, stream>>>(W2, W2T, D4, D);
  }

  // -------- 2. LN1 -> xn (bf16) --------
  k_layernorm<<<Mr, 256, 0, stream>>>(x, g1, be1, nullptr, xn, D);

  // -------- 3. Q/K/V projections (WMMA, bf16 outputs) --------
  {
    dim3 grid(D / 256, Mr / 16);
    k_gemm_bf16<<<grid, 128, 0, stream>>>(xn, WqT, nullptr, nullptr, nullptr, Qbuf, Mr, D, D, 0);
    k_gemm_bf16<<<grid, 128, 0, stream>>>(xn, WkT, nullptr, nullptr, nullptr, Kbuf, Mr, D, D, 0);
    k_gemm_bf16<<<grid, 128, 0, stream>>>(xn, WvT, nullptr, nullptr, nullptr, Vbuf, Mr, D, D, 0);
  }

  // -------- 4. memory K/V from h (tiny) --------
  k_rowmat<<<(Bn * D + 255) / 256, 256, 0, stream>>>(h, Wmk, Kmf, Bn, DM, D);
  k_rowmat<<<(Bn * D + 255) / 256, 256, 0, stream>>>(h, Wmv, Vmf, Bn, DM, D);

  // -------- 5. WMMA flash attention -> ao (bf16) --------
  {
    dim3 grid(S / 64, H, Bn);
    k_attention_wmma<<<grid, 128, 0, stream>>>(Qbuf, Kbuf, Vbuf, Kmf, Vmf, aob,
                                               Bn, S, H, DH);
  }

  // -------- 6. attended = x + ao @ Wo (WMMA, fused residual) --------
  {
    dim3 grid(D / 256, Mr / 16);
    k_gemm_bf16<<<grid, 128, 0, stream>>>(aob, WoT, nullptr, x, att, nullptr, Mr, D, D, 0);
  }

  // -------- 7. mean-pool + LSTM --------
  k_meanpool<<<(Bn * D + 255) / 256, 256, 0, stream>>>(att, pooled, Bn, S, D);
  k_gates<<<(Bn * D4 + 255) / 256, 256, 0, stream>>>(pooled, h, Wih, Whh, b_ih, b_hh, gates, Bn, D, D4);
  k_lstm<<<(Bn * DM + 255) / 256, 256, 0, stream>>>(gates, c, out_h, out_c, Bn, DM);

  // -------- 8. mixed = attended + (new_h @ Wmp) broadcast --------
  k_rowmat<<<(Bn * D + 255) / 256, 256, 0, stream>>>(out_h, Wmp, hrow, Bn, DM, D);
  {
    size_t tot = (size_t)Mr * D;
    k_addrow<<<(unsigned)((tot + 255) / 256), 256, 0, stream>>>(att, hrow, mixed, Bn, S, D);
  }

  // -------- 9. LN2 -> hn (bf16) --------
  k_layernorm<<<Mr, 256, 0, stream>>>(mixed, g2, be2, nullptr, hn, D);

  // -------- 10. FFN (WMMA): gelu(hn@W1+bf1) @ W2 + bf2 + mixed --------
  {
    dim3 grid1(D4 / 256, Mr / 16);
    k_gemm_bf16<<<grid1, 128, 0, stream>>>(hn, W1T, bf1, nullptr, nullptr, ffn1, Mr, D4, D, 1);
    dim3 grid2(D / 256, Mr / 16);
    k_gemm_bf16<<<grid2, 128, 0, stream>>>(ffn1, W2T, bf2, mixed, y, nullptr, Mr, D, D4, 0);
  }

  // -------- 11. LN3 -> out --------
  k_layernorm<<<Mr, 256, 0, stream>>>(y, g3, be3, out_tok, nullptr, D);
}
